// _MultiHeadSelfAttentionDP_23467701305779
// MI455X (gfx1250) — matrix operand order repaired and checked by reading, not compiled, not run on hardware
//
#include <hip/hip_runtime.h>
#include <hip/hip_bf16.h>
#include <math.h>

// ---------------------------------------------------------------------------
// MultiHeadSelfAttention for B=4, L=2048, D=512, H=8, hd=64 on gfx1250.
// bf16 WMMA (v_wmma_f32_16x16x32_bf16) for all four GEMMs + flash attention.
// Async global->LDS (ASYNCcnt) staging with double-buffered GEMM pipeline.
// Mask is bit-packed once (u32 per 32 keys) so the flash inner loop does two
// L2-resident u32 loads per row per key tile instead of 32 byte loads.
// ---------------------------------------------------------------------------

#define B_     4
#define L_     2048
#define D_     512
#define H_     8
#define HD_    64
#define M_     (B_ * L_)          // 8192 rows
#define SCALE_ 0.125f             // 1/sqrt(64)

typedef __attribute__((ext_vector_type(16))) __bf16 v16bf;
typedef __attribute__((ext_vector_type(8)))  float  v8f;
typedef int v4i __attribute__((vector_size(16))); // matches async builtin param

#if defined(__has_builtin)
#if __has_builtin(__builtin_amdgcn_global_load_async_to_lds_b128)
#define ASYNC_OK 1
#endif
#endif

#ifdef ASYNC_OK
// one 16-byte global->LDS async copy per lane (tracked by ASYNCcnt)
#define ASYNC_CP16(gsrc, ldst)                                                \
    __builtin_amdgcn_global_load_async_to_lds_b128(                           \
        (__attribute__((address_space(1))) v4i*)(gsrc),                       \
        (__attribute__((address_space(3))) v4i*)(ldst), 0, 0)
static __device__ __forceinline__ void wait_async0() {
#if __has_builtin(__builtin_amdgcn_s_wait_asynccnt)
    __builtin_amdgcn_s_wait_asynccnt(0);
#else
    asm volatile("s_wait_asynccnt 0" ::: "memory");
#endif
}
#else
#define ASYNC_CP16(gsrc, ldst) (*(uint4*)(ldst) = *(const uint4*)(gsrc))
static __device__ __forceinline__ void wait_async0() {}
#endif

union BFrag {
    v16bf          v;
    uint4          q[2];
    unsigned short u[16];
};
union U4U {
    uint4          q;
    unsigned short u[8];
    unsigned int   w[4];
};

static __device__ __forceinline__ unsigned short f2bf(float f) {
    unsigned int u = __float_as_uint(f);
    return (unsigned short)((u + 0x7FFFu + ((u >> 16) & 1u)) >> 16); // RNE
}
static __device__ __forceinline__ float bf2f(unsigned short u) {
    return __uint_as_float(((unsigned int)u) << 16);
}

static __device__ __forceinline__ v8f wmma_bf16(const BFrag& a, const BFrag& b, v8f c) {
    return __builtin_amdgcn_wmma_f32_16x16x32_bf16(
        false, a.v, false, b.v, (short)0, c, false, false);
}

// ---------------------------------------------------------------------------
// Kernel 1: fp32 -> bf16 (raw ushort) conversion
// ---------------------------------------------------------------------------
__global__ void cvt_f32_bf16(const float* __restrict__ s, unsigned short* __restrict__ d, int n) {
    for (int i = blockIdx.x * blockDim.x + threadIdx.x; i < n; i += gridDim.x * blockDim.x)
        d[i] = f2bf(s[i]);
}

// ---------------------------------------------------------------------------
// Kernel 1b: bit-pack boolean mask: 32 bytes -> one u32 (bit k = byte k != 0)
// ---------------------------------------------------------------------------
__global__ void mask_to_bits(const unsigned char* __restrict__ m,
                             unsigned int* __restrict__ bits, int nwords) {
    for (int i = blockIdx.x * blockDim.x + threadIdx.x; i < nwords;
         i += gridDim.x * blockDim.x) {
        const unsigned int* p = (const unsigned int*)(m + (size_t)i * 32);
        unsigned int w = 0;
#pragma unroll
        for (int k = 0; k < 8; ++k) {
            unsigned int u = p[k];
            if (u & 0x000000FFu) w |= 1u << (k * 4 + 0);
            if (u & 0x0000FF00u) w |= 1u << (k * 4 + 1);
            if (u & 0x00FF0000u) w |= 1u << (k * 4 + 2);
            if (u & 0xFF000000u) w |= 1u << (k * 4 + 3);
        }
        bits[i] = w;
    }
}

// ---------------------------------------------------------------------------
// Kernel 2/4: C = A(M x 512) * W^T(512 x 512) + bias.
// Double-buffered async global->LDS staging; 8 waves; BM=BN=128, BK=32.
// SPLIT=true : bf16 output, split heads into [B, H, L, 64]
// SPLIT=false: fp32 output, flat [M, 512]
// ---------------------------------------------------------------------------
template <bool SPLIT>
__global__ __launch_bounds__(256) void gemm_bf16_kernel(
    const unsigned short* __restrict__ A, const unsigned short* __restrict__ W,
    const float* __restrict__ bias, unsigned short* __restrict__ out_bf,
    float* __restrict__ out_f) {

    __shared__ __align__(16) unsigned short sA[2][128 * 48]; // 32 cols + pad (96B rows)
    __shared__ __align__(16) unsigned short sW[2][128 * 48];

    const int tid  = threadIdx.x;
    const int wid  = tid >> 5;
    const int lane = tid & 31;
    const int bm   = blockIdx.x * 128;
    const int bn   = blockIdx.y * 128;
    const int wm   = (wid & 3) * 32;
    const int wn   = (wid >> 2) * 64;
    const int l15  = lane & 15;
    const int lhi  = lane >> 4;
    const int ldr  = tid >> 2;          // staging row (0..63, x2 chunks)
    const int ldc  = (tid & 3) * 8;     // staging col in elements

    v8f acc[2][4];
#pragma unroll
    for (int mi = 0; mi < 2; ++mi)
#pragma unroll
        for (int ni = 0; ni < 4; ++ni) acc[mi][ni] = (v8f){0,0,0,0,0,0,0,0};

#define GEMM_STAGE(kt, bsel)                                                          \
    do {                                                                              \
        const int k0_ = (kt) * 32;                                                    \
        _Pragma("unroll")                                                             \
        for (int i_ = 0; i_ < 2; ++i_) {                                              \
            int r_ = ldr + i_ * 64;                                                   \
            ASYNC_CP16(A + (size_t)(bm + r_) * 512 + k0_ + ldc,                       \
                       &sA[bsel][r_ * 48 + ldc]);                                     \
            ASYNC_CP16(W + (size_t)(bn + r_) * 512 + k0_ + ldc,                       \
                       &sW[bsel][r_ * 48 + ldc]);                                     \
        }                                                                             \
    } while (0)

    GEMM_STAGE(0, 0);
    wait_async0();
    __syncthreads();

    for (int kt = 0; kt < 16; ++kt) {
        const int cur = kt & 1;
        if (kt < 15) GEMM_STAGE(kt + 1, cur ^ 1);  // stream next tile during compute

        const unsigned short* cA = sA[cur];
        const unsigned short* cW = sW[cur];
        // A-frags: lane = M row, halves = K {kb..kb+7, kb+16..kb+23}
        BFrag af[2];
#pragma unroll
        for (int mi = 0; mi < 2; ++mi) {
            int row = wm + mi * 16 + l15;
            int kb  = lhi * 8;
            af[mi].q[0] = *(const uint4*)(&cA[row * 48 + kb]);
            af[mi].q[1] = *(const uint4*)(&cA[row * 48 + kb + 16]);
        }
        // B-frags: lane = N col, halves = K {kb..kb+7, kb+16..kb+23} (matches A)
        BFrag bfr[4];
#pragma unroll
        for (int ni = 0; ni < 4; ++ni) {
            int col = wn + ni * 16 + l15;
            int kb  = lhi * 8;
            bfr[ni].q[0] = *(const uint4*)(&cW[col * 48 + kb]);
            bfr[ni].q[1] = *(const uint4*)(&cW[col * 48 + kb + 16]);
        }
#pragma unroll
        for (int mi = 0; mi < 2; ++mi)
#pragma unroll
            for (int ni = 0; ni < 4; ++ni)
                acc[mi][ni] = wmma_bf16(af[mi], bfr[ni], acc[mi][ni]);

        if (kt < 15) wait_async0();
        __syncthreads();
    }

    // epilogue: bias + store (C layout: lane = col, rows lhi*8+v)
#pragma unroll
    for (int mi = 0; mi < 2; ++mi)
#pragma unroll
        for (int ni = 0; ni < 4; ++ni) {
            const int col = bn + wn + ni * 16 + l15;
            const float bv = bias[col];
#pragma unroll
            for (int v = 0; v < 8; ++v) {
                const int row = bm + wm + mi * 16 + lhi * 8 + v;
                float val = acc[mi][ni][v] + bv;
                if (SPLIT) {
                    int b = row >> 11, l = row & 2047, h = col >> 6, d = col & 63;
                    out_bf[((((size_t)b * H_ + h) * L_ + l) << 6) + d] = f2bf(val);
                } else {
                    out_f[(size_t)row * D_ + col] = val;
                }
            }
        }
#undef GEMM_STAGE
}

// ---------------------------------------------------------------------------
// Kernel 3: flash attention. Q,K,V bf16 [B,H,L,64]. Block = 128 thr (4 waves),
// 64 query rows per block (16 per wave). Online softmax, key tiles of 64.
// K tile staged via async global->LDS; V staged TRANSPOSED so PV B-fragments
// are contiguous uint4 LDS loads. Mask read as bit-packed u32 words.
// ---------------------------------------------------------------------------
__global__ __launch_bounds__(128) void flash_attn_kernel(
    const unsigned short* __restrict__ Qp, const unsigned short* __restrict__ Kp,
    const unsigned short* __restrict__ Vp, const unsigned int* __restrict__ mbits,
    unsigned short* __restrict__ Om) {

    __shared__ __align__(16) unsigned short sQP[64 * 64]; // Q staging, then P
    __shared__ __align__(16) unsigned short sK [64 * 72];
    __shared__ __align__(16) unsigned short sVt[64 * 72]; // transposed: [d][key]

    const int tid  = threadIdx.x;
    const int wid  = tid >> 5;
    const int lane = tid & 31;
    const int l15  = lane & 15;
    const int lhi  = lane >> 4;
    const int bh   = blockIdx.y;
    const int b    = bh >> 3;
    const int h    = bh & 7;
    const int q0   = blockIdx.x * 64;
    const size_t headBase = (size_t)bh * L_ * HD_;

    // ---- stage Q tile (async), build A-frags (16 rows per wave, K=64) ----
#pragma unroll
    for (int i = 0; i < 4; ++i) {
        int idx = tid + i * 128;
        int r = idx >> 3, c = (idx & 7) * 8;
        ASYNC_CP16(Qp + headBase + (size_t)(q0 + r) * HD_ + c, &sQP[r * 64 + c]);
    }
    wait_async0();
    __syncthreads();
    const int qrl = wid * 16 + l15;
    const int kb8 = lhi * 8;
    BFrag qf[2];
    qf[0].q[0] = *(const uint4*)(&sQP[qrl * 64 + kb8]);
    qf[0].q[1] = *(const uint4*)(&sQP[qrl * 64 + kb8 + 16]);
    qf[1].q[0] = *(const uint4*)(&sQP[qrl * 64 + 32 + kb8]);
    qf[1].q[1] = *(const uint4*)(&sQP[qrl * 64 + 32 + kb8 + 16]);
    __syncthreads(); // sQP now reusable for P

    v8f o[4];
#pragma unroll
    for (int ni = 0; ni < 4; ++ni) o[ni] = (v8f){0,0,0,0,0,0,0,0};
    float mrow[8], lrow[8];
#pragma unroll
    for (int v = 0; v < 8; ++v) { mrow[v] = -__builtin_inff(); lrow[v] = 0.0f; }
    const int rbase = lhi * 8;
    const int kb16  = lhi * 8;
    // per-row base pointer into bit-packed mask (64 words per query row)
    const unsigned int* mrow0 = mbits + ((size_t)b * L_ + q0 + wid * 16 + rbase) * (L_ / 32);

    for (int jt = 0; jt < L_ / 64; ++jt) {
        const int kk0 = jt * 64;
        // ---- stage K (async) and V transposed (manual) ----
#pragma unroll
        for (int i = 0; i < 4; ++i) {
            int idx = tid + i * 128;
            int r = idx >> 3, c = (idx & 7) * 8;
            ASYNC_CP16(Kp + headBase + (size_t)(kk0 + r) * HD_ + c, &sK[r * 72 + c]);
        }
#pragma unroll
        for (int i = 0; i < 4; ++i) { // V: global uint4 -> 8 transposed b16 stores
            int idx = tid + i * 128;
            int r = idx >> 3, c = (idx & 7) * 8;
            U4U t;
            t.q = *(const uint4*)(Vp + headBase + (size_t)(kk0 + r) * HD_ + c);
#pragma unroll
            for (int e = 0; e < 8; ++e) sVt[(c + e) * 72 + r] = t.u[e];
        }
        wait_async0();
        __syncthreads();

        // ---- S = Q K^T : B-frag lane col = key, K dim = d (contiguous) ----
        v8f s[4];
#pragma unroll
        for (int ni = 0; ni < 4; ++ni) {
            const int krow = ni * 16 + l15;
            BFrag bk0, bk1;
            bk0.q[0] = *(const uint4*)(&sK[krow * 72 + kb16]);
            bk0.q[1] = *(const uint4*)(&sK[krow * 72 + kb16 + 16]);
            bk1.q[0] = *(const uint4*)(&sK[krow * 72 + 32 + kb16]);
            bk1.q[1] = *(const uint4*)(&sK[krow * 72 + 32 + kb16 + 16]);
            s[ni] = (v8f){0,0,0,0,0,0,0,0};
            s[ni] = wmma_bf16(qf[0], bk0, s[ni]);
            s[ni] = wmma_bf16(qf[1], bk1, s[ni]);
        }

        // ---- mask (bit-packed) + online softmax (rows in 16-lane groups) ----
#pragma unroll
        for (int v = 0; v < 8; ++v) {
            const int rl = wid * 16 + rbase + v; // row within block tile
            const unsigned int w0 = mrow0[v * (L_ / 32) + jt * 2];     // keys kk0..kk0+31
            const unsigned int w1 = mrow0[v * (L_ / 32) + jt * 2 + 1]; // keys kk0+32..+63
            float sv[4];
#pragma unroll
            for (int ni = 0; ni < 4; ++ni) {
                const unsigned int w  = (ni < 2) ? w0 : w1;
                const int          sh = l15 + (ni & 1) * 16;
                float x = s[ni][v] * SCALE_;
                if ((w >> sh) & 1u) x = -__builtin_inff();
                sv[ni] = x;
            }
            float rv = fmaxf(fmaxf(sv[0], sv[1]), fmaxf(sv[2], sv[3]));
#pragma unroll
            for (int off = 1; off < 16; off <<= 1)
                rv = fmaxf(rv, __shfl_xor(rv, off, 32));
            const float nm   = fmaxf(mrow[v], rv);
            const bool  live = (nm != -__builtin_inff());
            const float alpha = live ? __expf(mrow[v] - nm) : 1.0f;
            float rsum = 0.0f;
#pragma unroll
            for (int ni = 0; ni < 4; ++ni) {
                float p = live ? __expf(sv[ni] - nm) : 0.0f;
                rsum += p;
                sQP[rl * 64 + ni * 16 + l15] = f2bf(p);
            }
#pragma unroll
            for (int off = 1; off < 16; off <<= 1)
                rsum += __shfl_xor(rsum, off, 32);
            lrow[v] = lrow[v] * alpha + rsum;
            mrow[v] = nm;
#pragma unroll
            for (int ni = 0; ni < 4; ++ni) o[ni][v] *= alpha;
        }

        // ---- O += P V : P A-frags from LDS; V B-frags contiguous from sVt ----
        BFrag pf0, pf1;
        pf0.q[0] = *(const uint4*)(&sQP[qrl * 64 + kb8]);
        pf0.q[1] = *(const uint4*)(&sQP[qrl * 64 + kb8 + 16]);
        pf1.q[0] = *(const uint4*)(&sQP[qrl * 64 + 32 + kb8]);
        pf1.q[1] = *(const uint4*)(&sQP[qrl * 64 + 32 + kb8 + 16]);
#pragma unroll
        for (int ni = 0; ni < 4; ++ni) {
            const int dcol = ni * 16 + l15;
            BFrag bv0, bv1; // B[k=key][n=d] = sVt[d][key]: contiguous in key
            bv0.q[0] = *(const uint4*)(&sVt[dcol * 72 + kb16]);
            bv0.q[1] = *(const uint4*)(&sVt[dcol * 72 + kb16 + 16]);
            bv1.q[0] = *(const uint4*)(&sVt[dcol * 72 + 32 + kb16]);
            bv1.q[1] = *(const uint4*)(&sVt[dcol * 72 + 32 + kb16 + 16]);
            o[ni] = wmma_bf16(pf0, bv0, o[ni]);
            o[ni] = wmma_bf16(pf1, bv1, o[ni]);
        }
        __syncthreads();
    }

    // ---- fully-masked fix-up (attend to self), normalize, store merged ----
#pragma unroll
    for (int v = 0; v < 8; ++v) {
        const int qg = q0 + wid * 16 + rbase + v;
        const bool dead = (mrow[v] == -__builtin_inff());
        const float inv = dead ? 1.0f : (1.0f / lrow[v]);
#pragma unroll
        for (int ni = 0; ni < 4; ++ni) {
            const int d = ni * 16 + l15;
            float ov = o[ni][v];
            if (dead) ov = bf2f(Vp[headBase + (size_t)qg * HD_ + d]); // weight 1 on self
            Om[((size_t)b * L_ + qg) * D_ + h * HD_ + d] = f2bf(ov * inv);
        }
    }
}

// ---------------------------------------------------------------------------
extern "C" void kernel_launch(void* const* d_in, const int* in_sizes, int n_in,
                              void* d_out, int out_size, void* d_ws, size_t ws_size,
                              hipStream_t stream) {
    const float* x  = (const float*)d_in[0];
    // numpy/jax bool array: 1 byte per element, nonzero == masked-out
    const unsigned char* attn_mask = (const unsigned char*)d_in[1];
    const float* Wq = (const float*)d_in[2];
    const float* bq = (const float*)d_in[3];
    const float* Wk = (const float*)d_in[4];
    const float* bk = (const float*)d_in[5];
    const float* Wv = (const float*)d_in[6];
    const float* bv = (const float*)d_in[7];
    const float* Wo = (const float*)d_in[8];
    const float* bo = (const float*)d_in[9];

    // workspace carve-up (bf16 raw ushorts, then u32 mask bits)
    unsigned short* Xb  = (unsigned short*)d_ws;          // 8192*512
    unsigned short* Wqb = Xb  + (size_t)M_ * D_;
    unsigned short* Wkb = Wqb + (size_t)D_ * D_;
    unsigned short* Wvb = Wkb + (size_t)D_ * D_;
    unsigned short* Wob = Wvb + (size_t)D_ * D_;
    unsigned short* Qh  = Wob + (size_t)D_ * D_;          // [B,H,L,64] each
    unsigned short* Kh  = Qh  + (size_t)M_ * D_;
    unsigned short* Vh  = Kh  + (size_t)M_ * D_;
    unsigned short* Omg = Vh  + (size_t)M_ * D_;          // merged heads [M, 512]
    unsigned int*   Mb  = (unsigned int*)(Omg + (size_t)M_ * D_); // [B, L, L/32]

    // 1) fp32 -> bf16 + mask bit-pack
    cvt_f32_bf16<<<2048, 256, 0, stream>>>(x,  Xb,  M_ * D_);
    cvt_f32_bf16<<<512,  256, 0, stream>>>(Wq, Wqb, D_ * D_);
    cvt_f32_bf16<<<512,  256, 0, stream>>>(Wk, Wkb, D_ * D_);
    cvt_f32_bf16<<<512,  256, 0, stream>>>(Wv, Wvb, D_ * D_);
    cvt_f32_bf16<<<512,  256, 0, stream>>>(Wo, Wob, D_ * D_);
    mask_to_bits<<<1024, 256, 0, stream>>>(attn_mask, Mb, B_ * L_ * (L_ / 32));

    // 2) QKV projections (fused bias + head split)
    dim3 ggrid(M_ / 128, D_ / 128);
    gemm_bf16_kernel<true><<<ggrid, 256, 0, stream>>>(Xb, Wqb, bq, Qh, nullptr);
    gemm_bf16_kernel<true><<<ggrid, 256, 0, stream>>>(Xb, Wkb, bk, Kh, nullptr);
    gemm_bf16_kernel<true><<<ggrid, 256, 0, stream>>>(Xb, Wvb, bv, Vh, nullptr);

    // 3) flash attention -> merged-head bf16
    dim3 fgrid(L_ / 64, B_ * H_);
    flash_attn_kernel<<<fgrid, 128, 0, stream>>>(Qh, Kh, Vh, Mb, Omg);

    // 4) output projection, fp32 result
    gemm_bf16_kernel<false><<<ggrid, 256, 0, stream>>>(Omg, Wob, bo, nullptr, (float*)d_out);
}
